// LorentzPureConv1d_21199958573237
// MI455X (gfx1250) — compile-verified, matrix operand order checked by reading, not run on hardware
//
#include <hip/hip_runtime.h>

typedef __attribute__((ext_vector_type(16))) __bf16 v16bf;
typedef __attribute__((ext_vector_type(8)))  __bf16 v8bf;
typedef __attribute__((ext_vector_type(8)))  float  v8f;
typedef __attribute__((ext_vector_type(4)))  int    v4i;

// Problem constants
#define BATCH 16
#define LEN   8192
#define CH    257      // input channels (0 = time, 1..256 = space)
#define CS    256      // space channels
#define NF    4096     // frames per batch
#define OC    257      // output channels
#define LIN   1281     // W row length (0 = time column)
#define PD    2
#define BM    64       // frames per block
#define XROWS 131      // input rows staged per block: 2*63 + 4 + 1
#define XSTB  264      // bf16 LDS row stride (16B aligned, de-conflicted)
#define OST   268      // f32 epilogue stage stride (de-conflicted)
#define KSTEPS 40      // 1280 / 32
#define NPANEL 20      // 2 k-steps per panel
#define PANEL  32768   // bytes per B panel: 64k x 256n bf16

#define XSB_BYTES (XROWS * XSTB * 2)        // 69168
#define RSQ_OFF   XSB_BYTES                 // rowsq (131 floats)
#define BBUF_OFF  (RSQ_OFF + 528)           // 69696, 16B aligned
#define SMEM_BYTES (BBUF_OFF + 2 * PANEL)   // 135232

#if defined(__gfx1250__) && \
    __has_builtin(__builtin_amdgcn_global_load_async_to_lds_b128) && \
    __has_builtin(__builtin_amdgcn_s_wait_asynccnt)
#define USE_ASYNC 1
#else
#define USE_ASYNC 0
#endif

// ---------------------------------------------------------------------------
// Pack W into bf16 in WMMA-B-fragment order:
// layout [ks][jtile][lane][16 halves]:
//   kbase = (l<16?0:8); K32 = kbase + (h<8 ? h : h+8); kg = ks*32+K32
//   o = jtile*16 + (l&15) + 1 ; feature = 1 + kg
// ---------------------------------------------------------------------------
__global__ __launch_bounds__(256)
void prep_wb(const float* __restrict__ W, __bf16* __restrict__ Wb) {
    int idx = blockIdx.x * 256 + threadIdx.x;          // < 40*16*32*16 = 327680
    int h  = idx & 15;
    int l  = (idx >> 4) & 31;
    int j  = (idx >> 9) & 15;
    int ks = idx >> 13;
    int K32 = ((l >> 4) << 3) + (h < 8 ? h : h + 8);
    int kg  = ks * 32 + K32;                           // 0..1279
    int o   = j * 16 + (l & 15) + 1;                   // 1..256
    Wb[idx] = (__bf16)W[(long)o * LIN + 1 + kg];
}

#if USE_ASYNC
__device__ __forceinline__
void stage_b_panel(int p, int tid, const __bf16* Wb, char* bbuf) {
    typedef __attribute__((address_space(1))) v4i* gp_t;   // "__device__ v4i*"
    typedef __attribute__((address_space(3))) v4i* lp_t;   // "__shared__ v4i*"
    gp_t g = (gp_t)((const char*)Wb + p * PANEL + tid * 16);
    lp_t l = (lp_t)(bbuf + (p & 1) * PANEL + tid * 16);
    #pragma unroll
    for (int i = 0; i < 8; ++i)
        __builtin_amdgcn_global_load_async_to_lds_b128(g + i * 256,
                                                       l + i * 256, 0, 0);
}
#endif

// ---------------------------------------------------------------------------
// Fused kernel: bf16 stage of x (+ exact f32 row sumsq) -> bf16 WMMA GEMM
// (wave tile 32x64, async double-buffered W panels) -> f32 Lorentz epilogue.
// ---------------------------------------------------------------------------
__global__ __launch_bounds__(256)
void lorentz_gemm(const float* __restrict__ x, const float* __restrict__ W,
                  const float* __restrict__ bias, const __bf16* __restrict__ Wb,
                  float* __restrict__ out) {
    __shared__ __align__(16) char smem[SMEM_BYTES];
    __bf16* xsb  = (__bf16*)smem;
    float* rowsq = (float*)(smem + RSQ_OFF);
    char*  bbuf  = smem + BBUF_OFF;

    const int tid  = threadIdx.x;
    const int lane = tid & 31;
    const int wid  = tid >> 5;
    const int bid  = blockIdx.x;

    const int b  = (bid * BM) / NF;
    const int n0 = (bid * BM) % NF;
    const long xbase = (long)b * LEN * CH;
    const int  t0    = 2 * n0 - PD;

    // ---- fused stage (f32 -> bf16 in LDS) + exact f32 row sum-of-squares ----
    for (int tr = wid; tr < XROWS; tr += 8) {
        int tg = t0 + tr;
        float v[8];
        if (tg >= 0 && tg < LEN) {
            const float* src = &x[xbase + (long)tg * CH + 1 + lane * 8];
            #pragma unroll
            for (int i = 0; i < 8; ++i) v[i] = src[i];
        } else {
            #pragma unroll
            for (int i = 0; i < 8; ++i) v[i] = 0.0f;
        }
        float sq = 0.0f;
        v8bf hv;
        #pragma unroll
        for (int i = 0; i < 8; ++i) { sq += v[i] * v[i]; hv[i] = (__bf16)v[i]; }
        *(v8bf*)&xsb[tr * XSTB + lane * 8] = hv;
        #pragma unroll
        for (int off = 16; off; off >>= 1) sq += __shfl_xor(sq, off, 32);
        if (lane == 0) rowsq[tr] = sq;
    }
    __syncthreads();

    // ---- bf16 WMMA main loop: wave tile = 32 rows x 64 cols (2x4 tiles) ----
    const int wn = wid & 3;                  // column group of 64
    const int wm = wid >> 2;                 // row group of 32
    const int ml = lane & 15;
    const int khalf = lane >> 4;             // A-layout K-half select
    const int baserow = 2 * (wm * 32 + ml);  // second A tile at +32
    const int coff = khalf * 8;

    v8f acc[2][4] = {};

#if USE_ASYNC
    stage_b_panel(0, tid, Wb, bbuf);
    stage_b_panel(1, tid, Wb, bbuf);
#endif

    for (int p = 0; p < NPANEL; ++p) {
#if USE_ASYNC
        if (p < NPANEL - 1) __builtin_amdgcn_s_wait_asynccnt(8);
        else                __builtin_amdgcn_s_wait_asynccnt(0);
        __syncthreads();                     // panel p visible to all waves
#endif
        #pragma unroll
        for (int sub = 0; sub < 2; ++sub) {
            const int ks  = p * 2 + sub;
            const int tap = ks >> 3;         // 1280 ordered as (tap, channel)
            const int c0  = (ks & 7) << 5;

            v16bf af[2];
            #pragma unroll
            for (int mi = 0; mi < 2; ++mi) {
                const __bf16* xr =
                    &xsb[(baserow + 32 * mi + tap) * XSTB + c0 + coff];
                v8bf lo = *(const v8bf*)xr;          // K kb+0..7
                v8bf hi = *(const v8bf*)(xr + 16);   // K kb+16..23
                af[mi] = __builtin_shufflevector(lo, hi,
                             0, 1, 2, 3, 4, 5, 6, 7,
                             8, 9, 10, 11, 12, 13, 14, 15);
            }
#if USE_ASYNC
            const char* bp = bbuf + (p & 1) * PANEL + sub * (PANEL / 2);
            #pragma unroll
            for (int j = 0; j < 4; ++j) {
                v16bf bfrag =
                    *(const v16bf*)(bp + (((wn * 4 + j) * 32 + lane) << 5));
                acc[0][j] = __builtin_amdgcn_wmma_f32_16x16x32_bf16(
                    false, af[0], false, bfrag, (short)0, acc[0][j], false, false);
                acc[1][j] = __builtin_amdgcn_wmma_f32_16x16x32_bf16(
                    false, af[1], false, bfrag, (short)0, acc[1][j], false, false);
            }
#else
            const v16bf* wbp =
                ((const v16bf*)Wb) + ((ks * 16 + wn * 4) * 32 + lane);
            #pragma unroll
            for (int j = 0; j < 4; ++j) {
                v16bf bfrag = wbp[j * 32];
                acc[0][j] = __builtin_amdgcn_wmma_f32_16x16x32_bf16(
                    false, af[0], false, bfrag, (short)0, acc[0][j], false, false);
                acc[1][j] = __builtin_amdgcn_wmma_f32_16x16x32_bf16(
                    false, af[1], false, bfrag, (short)0, acc[1][j], false, false);
            }
#endif
        }
#if USE_ASYNC
        __syncthreads();                     // all reads of this buffer done
        if (p + 2 < NPANEL) stage_b_panel(p + 2, tid, Wb, bbuf);
#endif
    }
    __syncthreads();                         // done with xsb; reuse as outstage

    // ---- dump accumulators to LDS (C layout: VGPR r -> row r / r+8) ----
    float* outst = (float*)smem;             // 64 rows x OST stride (fits xsb)
    #pragma unroll
    for (int mi = 0; mi < 2; ++mi) {
        int rowbase = wm * 32 + mi * 16 + khalf * 8;
        #pragma unroll
        for (int j = 0; j < 4; ++j) {
            int col = (wn * 4 + j) * 16 + ml;
            #pragma unroll
            for (int r = 0; r < 8; ++r)
                outst[(rowbase + r) * OST + col] = acc[mi][j][r];
        }
    }
    __syncthreads();

    // ---- epilogue: +time*W[:,0] + bias, clamp norm, Lorentz time channel ----
    for (int row = wid; row < BM; row += 8) {
        float t2 = 1.0f;
        #pragma unroll
        for (int tap = 0; tap < 5; ++tap) t2 += rowsq[2 * row + tap];
        float tval = sqrtf(t2);

        float sv[8];
        float n2 = 0.0f;
        #pragma unroll
        for (int i = 0; i < 8; ++i) {
            int c = lane + i * 32;
            int o = c + 1;
            float s = outst[row * OST + c] + tval * W[(long)o * LIN] + bias[o];
            sv[i] = s;
            n2 += s * s;
        }
        #pragma unroll
        for (int off = 16; off; off >>= 1) n2 += __shfl_xor(n2, off, 32);

        float norm  = sqrtf(n2);
        float scale = fminf(1.0f, 1000.0f / fmaxf(norm, 1e-8f));
        long obase = ((long)b * NF + n0 + row) * OC;
        if (lane == 0) out[obase] = sqrtf(1.0f + scale * scale * n2);
        #pragma unroll
        for (int i = 0; i < 8; ++i)
            out[obase + 1 + lane + i * 32] = sv[i] * scale;
    }
}

extern "C" void kernel_launch(void* const* d_in, const int* in_sizes, int n_in,
                              void* d_out, int out_size, void* d_ws, size_t ws_size,
                              hipStream_t stream) {
    const float* x    = (const float*)d_in[0];
    const float* W    = (const float*)d_in[1];
    const float* bias = (const float*)d_in[2];
    __bf16* Wb = (__bf16*)d_ws;              // 655,360 bytes of scratch

    prep_wb<<<1280, 256, 0, stream>>>(W, Wb);

    const int mblocks = (BATCH * NF) / BM;   // 1024
    lorentz_gemm<<<mblocks, 256, 0, stream>>>(x, W, bias, Wb, (float*)d_out);
}
